// LlamaAttention_9165460210122
// MI455X (gfx1250) — matrix operand order repaired and checked by reading, not compiled, not run on hardware
//
#include <hip/hip_runtime.h>

// ---------------------------------------------------------------------------
// Problem constants (B=2, S=1024, H=2048, NH=32, NKV=8, HD=64)
// ---------------------------------------------------------------------------
#define TOK 2048          // B*S tokens
#define SEQ 1024
#define ATT_SCALE 0.125f  // HD^-0.5

typedef __attribute__((ext_vector_type(2)))  float  v2f;
typedef __attribute__((ext_vector_type(4)))  float  v4f;
typedef __attribute__((ext_vector_type(2)))  __bf16 v2bf;
typedef __attribute__((ext_vector_type(4)))  __bf16 v4bf;
typedef __attribute__((ext_vector_type(8)))  __bf16 v8bf;
typedef __attribute__((ext_vector_type(16))) __bf16 v16bf;
typedef __attribute__((ext_vector_type(8)))  float  v8f;

union V16u { v16bf v; v8bf h[2]; unsigned int d[8]; unsigned short u[16]; };
union V8u  { v8f v; float f[8]; };
union Pk4  { v4bf v; uint2 u2; };
union Pk2  { v2bf v; unsigned int u; };

// native f32 -> bf16 conversions (let the backend pick v_cvt_pk_bf16_f32)
__device__ __forceinline__ uint2 cvt4bf(v4f f) {
  Pk4 p; p.v = __builtin_convertvector(f, v4bf);
  return p.u2;
}
__device__ __forceinline__ unsigned int pack2bf(float a, float b) {
  v2f f; f.x = a; f.y = b;
  Pk2 p; p.v = __builtin_convertvector(f, v2bf);
  return p.u;
}
__device__ __forceinline__ unsigned short bf16bits(float a) {
  union { __bf16 b; unsigned short s; } x; x.b = (__bf16)a; return x.s;
}

// WMMA operand layouts (ISA 7.12.2):
//  A (16x32): lane half 0: K runs {0..7, 16..23}; half 1: {8..15, 24..31}
//  B (32x16): same order as A; this code uses contiguous K = 16*half + j for both A and B

// ---------------------------------------------------------------------------
// Generic GEMM: C[TOK rows, N] = A(f32) x B(f32) (+bias), bf16 WMMA, f32 acc.
// bTrans=1: B is (N,K) row-major (A @ B^T). bTrans=0: B is (K,N).
// Grid: (N/32, TOK/64), 128 threads = 4 waves; wave w: rows w*16..+15, 32 cols.
// ---------------------------------------------------------------------------
__global__ __launch_bounds__(128) void gemm_bf16_kernel(
    const float* __restrict__ A, const float* __restrict__ Bm,
    const float* __restrict__ bias, float* __restrict__ C,
    int N, int K, int lda, int ldb, int ldc, int bTrans)
{
  __shared__ unsigned short As[64][40];  // 64 rows x 32 K (+pad), 80B pitch
  __shared__ unsigned short Bt[32][40];  // n-major: 32 n x 32 K (+pad)
  const int t = threadIdx.x;
  const int w = t >> 5, lane = t & 31, hl = lane >> 4, l4 = lane & 15;
  const int row0 = blockIdx.y * 64;
  const int n0 = blockIdx.x * 32;

  V8u acc0, acc1;
#pragma unroll
  for (int r = 0; r < 8; ++r) { acc0.f[r] = 0.0f; acc1.f[r] = 0.0f; }

  for (int k0 = 0; k0 < K; k0 += 32) {
    // stage A: 64 x 8 float4, b128 global loads -> packed cvt -> b64 LDS stores
#pragma unroll
    for (int i = 0; i < 4; ++i) {
      int idx = t + i * 128, r = idx >> 3, c4 = idx & 7;
      const v4f f = *(const v4f*)&A[(size_t)(row0 + r) * lda + k0 + c4 * 4];
      *(uint2*)&As[r][c4 * 4] = cvt4bf(f);
    }
    // stage B n-major
    if (bTrans) {
#pragma unroll
      for (int i = 0; i < 2; ++i) {
        int idx = t + i * 128, n = idx >> 3, c4 = idx & 7;
        const v4f f = *(const v4f*)&Bm[(size_t)(n0 + n) * ldb + k0 + c4 * 4];
        *(uint2*)&Bt[n][c4 * 4] = cvt4bf(f);
      }
    } else {
#pragma unroll
      for (int i = 0; i < 2; ++i) {
        int idx = t + i * 128, kk = idx >> 3, n4 = idx & 7;
        const v4f f = *(const v4f*)&Bm[(size_t)(k0 + kk) * ldb + n0 + n4 * 4];
        Bt[n4 * 4 + 0][kk] = bf16bits(f.x);
        Bt[n4 * 4 + 1][kk] = bf16bits(f.y);
        Bt[n4 * 4 + 2][kk] = bf16bits(f.z);
        Bt[n4 * 4 + 3][kk] = bf16bits(f.w);
      }
    }
    if (k0 + 32 < K)
      __builtin_prefetch(&A[(size_t)(row0 + w * 16) * lda + (k0 + 32)], 0, 1);
    __syncthreads();

    // operand gathers: 2x ds_load_b128 each
    V16u a, b0, b1;
    const v8bf* Ar = (const v8bf*)&As[w * 16 + l4][0];
    a.h[0] = Ar[hl * 2];      // K = hl*16 .. +7
    a.h[1] = Ar[hl * 2 + 1];  // K = hl*16 + 8 .. +15
    const v8bf* Br0 = (const v8bf*)&Bt[l4][0];
    b0.h[0] = Br0[hl * 2];    // K = hl*16 .. +7
    b0.h[1] = Br0[hl * 2 + 1];
    const v8bf* Br1 = (const v8bf*)&Bt[16 + l4][0];
    b1.h[0] = Br1[hl * 2];
    b1.h[1] = Br1[hl * 2 + 1];

    acc0.v = __builtin_amdgcn_wmma_f32_16x16x32_bf16(
        false, a.v, false, b0.v, (short)0, acc0.v, false, false);
    acc1.v = __builtin_amdgcn_wmma_f32_16x16x32_bf16(
        false, a.v, false, b1.v, (short)0, acc1.v, false, false);
    __syncthreads();
  }

  const float bias0 = bias ? bias[n0 + l4] : 0.0f;
  const float bias1 = bias ? bias[n0 + 16 + l4] : 0.0f;
#pragma unroll
  for (int r = 0; r < 8; ++r) {
    int row = row0 + w * 16 + r + 8 * hl;
    C[(size_t)row * ldc + n0 + l4]      = acc0.f[r] + bias0;
    C[(size_t)row * ldc + n0 + 16 + l4] = acc1.f[r] + bias1;
  }
}

// ---------------------------------------------------------------------------
// Per-position 2x2 block rotation in low-rank space (in-place).
// ---------------------------------------------------------------------------
__global__ void rotate_kernel(float* __restrict__ x, const float* __restrict__ rot,
                              int total, int D) {
  int idx = blockIdx.x * blockDim.x + threadIdx.x;
  if (idx >= total) return;
  int pairs = D >> 1;
  int token = idx / pairs;
  int pr = idx - token * pairs;
  int s = token & (SEQ - 1);
  int hh = pr & 15;
  const float* r = rot + (((size_t)s * 16 + hh) << 2);
  float* p = x + (size_t)token * D + 2 * pr;
  float t0 = p[0], t1 = p[1];
  p[0] = t0 * r[0] + t1 * r[2];
  p[1] = t0 * r[1] + t1 * r[3];
}

// copy column range [col0, col0+ncols) from src to dst (same ld), TOK rows
__global__ void copycols_kernel(const float* __restrict__ src, float* __restrict__ dst,
                                int ld, int col0, int ncols) {
  int idx = blockIdx.x * blockDim.x + threadIdx.x;
  int total = TOK * ncols;
  if (idx >= total) return;
  int r = idx / ncols, c = idx - r * ncols;
  size_t off = (size_t)r * ld + col0 + c;
  dst[off] = src[off];
}

// ---------------------------------------------------------------------------
// Flash attention, causal, GQA (32 q-heads over 8 kv-heads), HD=64.
// One wave per 16-row Q tile per (b, head). V tile staged transposed+packed
// in LDS so the P@V B-operand is two ds_load_b128; P is packed cross-lane.
// ---------------------------------------------------------------------------
__global__ __launch_bounds__(32) void attn_kernel(
    const float* __restrict__ Q, const float* __restrict__ Kmat,
    const float* __restrict__ Vmat, float* __restrict__ O)
{
  __shared__ unsigned int VtP[64][20];  // [hd][key/2], 2 bf16 per dword, 80B pitch
  __shared__ unsigned int PlP[16][20];  // [m][key/2]
  const int lane = threadIdx.x, hl = lane >> 4, l4 = lane & 15;
  const int q0 = blockIdx.x * 16;
  const int h = blockIdx.y;
  const int b = blockIdx.z;
  const int hk = h >> 2;                                  // NH/NKV = 4
  const size_t qbase = ((size_t)b * SEQ + q0) * 2048 + (size_t)h * 64;
  const size_t kbase = ((size_t)b * SEQ) * 512 + (size_t)hk * 64;

  // Q tile as two A-operands (HD chunks 0-31, 32-63); row m = l4; b128 loads
  V16u qa[2];
  const float* Qrow = Q + qbase + (size_t)l4 * 2048;
#pragma unroll
  for (int cc = 0; cc < 2; ++cc) {
    const v4f a0 = *(const v4f*)&Qrow[cc * 32 + hl * 16];
    const v4f a1 = *(const v4f*)&Qrow[cc * 32 + hl * 16 + 4];
    const v4f a2 = *(const v4f*)&Qrow[cc * 32 + hl * 16 + 8];
    const v4f a3 = *(const v4f*)&Qrow[cc * 32 + hl * 16 + 12];
    uint2 p0 = cvt4bf(a0), p1 = cvt4bf(a1), p2 = cvt4bf(a2), p3 = cvt4bf(a3);
    qa[cc].d[0] = p0.x; qa[cc].d[1] = p0.y;
    qa[cc].d[2] = p1.x; qa[cc].d[3] = p1.y;
    qa[cc].d[4] = p2.x; qa[cc].d[5] = p2.y;
    qa[cc].d[6] = p3.x; qa[cc].d[7] = p3.y;
  }

  V8u acc[4];                 // 16x64 f32 output accumulator (4 N-tiles)
  float mrow[8], lrow[8];     // per-row running max / sum (rows r+8*hl)
#pragma unroll
  for (int r = 0; r < 8; ++r) {
    mrow[r] = -1e30f; lrow[r] = 0.0f;
#pragma unroll
    for (int t4 = 0; t4 < 4; ++t4) acc[t4].f[r] = 0.0f;
  }

  for (int kt = 0; kt < q0 + 16; kt += 32) {              // 32 keys / step
    // stage V tile (32 keys x 64 hd) transposed + key-pair packed into LDS:
    // coalesced float2 loads (lanes sweep hd), two b32 LDS stores per pair
#pragma unroll
    for (int kp = 0; kp < 16; ++kp) {
      const v2f ea = *(const v2f*)&Vmat[kbase + (size_t)(kt + 2 * kp) * 512 + lane * 2];
      const v2f eb = *(const v2f*)&Vmat[kbase + (size_t)(kt + 2 * kp + 1) * 512 + lane * 2];
      VtP[lane * 2][kp]     = pack2bf(ea.x, eb.x);
      VtP[lane * 2 + 1][kp] = pack2bf(ea.y, eb.y);
    }

    // scores = Q K^T : 2 key sub-tiles x 2 HD chunks = 4 WMMAs
    V8u sc[2];
#pragma unroll
    for (int s2 = 0; s2 < 2; ++s2) {
#pragma unroll
      for (int r = 0; r < 8; ++r) sc[s2].f[r] = 0.0f;
#pragma unroll
      for (int cc = 0; cc < 2; ++cc) {
        // B operand: n = key (l4), K = hd = cc*32 + hl*16 + j  (contiguous)
        const float* Kr = Kmat + kbase + (size_t)(kt + s2 * 16 + l4) * 512 + cc * 32 + hl * 16;
        uint2 p0 = cvt4bf(*(const v4f*)&Kr[0]);
        uint2 p1 = cvt4bf(*(const v4f*)&Kr[4]);
        uint2 p2 = cvt4bf(*(const v4f*)&Kr[8]);
        uint2 p3 = cvt4bf(*(const v4f*)&Kr[12]);
        V16u kb;
        kb.d[0] = p0.x; kb.d[1] = p0.y;
        kb.d[2] = p1.x; kb.d[3] = p1.y;
        kb.d[4] = p2.x; kb.d[5] = p2.y;
        kb.d[6] = p3.x; kb.d[7] = p3.y;
        sc[s2].v = __builtin_amdgcn_wmma_f32_16x16x32_bf16(
            false, qa[cc].v, false, kb.v, (short)0, sc[s2].v, false, false);
      }
    }

    // scale + causal mask + online softmax (rows live in one 16-lane half)
#pragma unroll
    for (int r = 0; r < 8; ++r) {
      int mg = q0 + r + 8 * hl;
      float v0 = sc[0].f[r] * ATT_SCALE; if (kt + l4 > mg)      v0 = -1e30f;
      float v1 = sc[1].f[r] * ATT_SCALE; if (kt + 16 + l4 > mg) v1 = -1e30f;
      float mx = fmaxf(v0, v1);
#pragma unroll
      for (int msk = 8; msk; msk >>= 1) mx = fmaxf(mx, __shfl_xor(mx, msk, 32));
      float mn = fmaxf(mrow[r], mx);
      float fc = __expf(mrow[r] - mn);
      mrow[r] = mn;
      float p0 = __expf(v0 - mn), p1 = __expf(v1 - mn);
      float rs = p0 + p1;
#pragma unroll
      for (int msk = 8; msk; msk >>= 1) rs += __shfl_xor(rs, msk, 32);
      lrow[r] = lrow[r] * fc + rs;
#pragma unroll
      for (int t4 = 0; t4 < 4; ++t4) acc[t4].f[r] *= fc;
      // pack adjacent-key pair cross-lane, even lanes store one dword each
      float n0p = __shfl_xor(p0, 1, 32);
      float n1p = __shfl_xor(p1, 1, 32);
      if (!(l4 & 1)) {
        PlP[r + 8 * hl][l4 >> 1]       = pack2bf(p0, n0p);  // keys l4, l4+1
        PlP[r + 8 * hl][8 + (l4 >> 1)] = pack2bf(p1, n1p);  // keys 16+l4, 17+l4
      }
    }
    __syncthreads();

    // reload P in A layout (2x ds_load_b128); acc += P @ V (4 HD N-tiles)
    V16u pa;
    const v8bf* Pr = (const v8bf*)&PlP[l4][0];
    pa.h[0] = Pr[hl * 2];    // keys hl*16 .. +7
    pa.h[1] = Pr[hl * 2 + 1]; // keys hl*16 + 8 .. +15
#pragma unroll
    for (int t4 = 0; t4 < 4; ++t4) {
      V16u vb;               // B operand: K = key = hl*16 + j, n = hd (l4)
      const v8bf* Vr = (const v8bf*)&VtP[t4 * 16 + l4][0];
      vb.h[0] = Vr[hl * 2];
      vb.h[1] = Vr[hl * 2 + 1];
      acc[t4].v = __builtin_amdgcn_wmma_f32_16x16x32_bf16(
          false, pa.v, false, vb.v, (short)0, acc[t4].v, false, false);
    }
    __syncthreads();
  }

#pragma unroll
  for (int r = 0; r < 8; ++r) {
    float inv = 1.0f / lrow[r];
#pragma unroll
    for (int t4 = 0; t4 < 4; ++t4)
      O[qbase + (size_t)(r + 8 * hl) * 2048 + t4 * 16 + l4] = acc[t4].f[r] * inv;
  }
}

// ---------------------------------------------------------------------------
// Host-side orchestration
// ---------------------------------------------------------------------------
extern "C" void kernel_launch(void* const* d_in, const int* in_sizes, int n_in,
                              void* d_out, int out_size, void* d_ws, size_t ws_size,
                              hipStream_t stream) {
  (void)in_sizes; (void)n_in; (void)out_size; (void)ws_size;
  const float* hs  = (const float*)d_in[0];
  const float* rot = (const float*)d_in[1];
  // d_in[2] = additive causal mask; causal is applied analytically in attn_kernel
  const float* Wqd = (const float*)d_in[3];
  const float* bqd = (const float*)d_in[4];
  const float* Wqu = (const float*)d_in[5];
  const float* bqu = (const float*)d_in[6];
  const float* Wkd = (const float*)d_in[7];
  const float* bkd = (const float*)d_in[8];
  const float* Wku = (const float*)d_in[9];
  const float* bku = (const float*)d_in[10];
  const float* Wvd = (const float*)d_in[11];
  const float* bvd = (const float*)d_in[12];
  const float* Wvu = (const float*)d_in[13];
  const float* bvu = (const float*)d_in[14];
  const float* Wo  = (const float*)d_in[15];
  const float* B_q = (const float*)d_in[16];
  const float* B_k = (const float*)d_in[17];
  const float* C_q = (const float*)d_in[18];
  const float* C_k = (const float*)d_in[19];

  float* ws = (float*)d_ws;
  float* qh = ws;                          // (TOK,1024)  reused for rotated qh
  float* kh = qh + (size_t)TOK * 1024;     // (TOK,256)   reused for rotated kh
  float* vh = kh + (size_t)TOK * 256;      // (TOK,256)
  float* tq = vh + (size_t)TOK * 256;      // (TOK,1024)  rotate temp q
  float* tk = tq + (size_t)TOK * 1024;     // (TOK,256)   rotate temp k
  float* qq = tk + (size_t)TOK * 256;      // (TOK,2048)  q up-proj
  float* kk = qq + (size_t)TOK * 2048;     // (TOK,512)   k up-proj
  float* vv = kk + (size_t)TOK * 512;      // (TOK,512)   v up-proj
  float* qf = vv + (size_t)TOK * 512;      // (TOK,2048)  q after C_q overwrite
  float* kf = qf + (size_t)TOK * 2048;     // (TOK,512)   k after C_k overwrite
  float* at = kf + (size_t)TOK * 512;      // (TOK,2048)  attention output

  auto gemm = [&](const float* A, const float* Bm, const float* bias, float* C,
                  int N, int K, int lda, int ldb, int ldc, int bt) {
    dim3 g(N / 32, TOK / 64);
    gemm_bf16_kernel<<<g, 128, 0, stream>>>(A, Bm, bias, C, N, K, lda, ldb, ldc, bt);
  };

  // down projections: x @ W^T + b
  gemm(hs, Wqd, bqd, qh, 1024, 2048, 2048, 2048, 1024, 1);
  gemm(hs, Wkd, bkd, kh,  256, 2048, 2048, 2048,  256, 1);
  gemm(hs, Wvd, bvd, vh,  256, 2048, 2048, 2048,  256, 1);

  // HLA rotate q: t = qh @ B_q^T ; rotate pairs ; qh = t @ B_q
  gemm(qh, B_q, nullptr, tq, 1024, 1024, 1024, 1024, 1024, 1);
  rotate_kernel<<<(TOK * 512 + 255) / 256, 256, 0, stream>>>(tq, rot, TOK * 512, 1024);
  gemm(tq, B_q, nullptr, qh, 1024, 1024, 1024, 1024, 1024, 0);

  // HLA rotate k
  gemm(kh, B_k, nullptr, tk, 256, 256, 256, 256, 256, 1);
  rotate_kernel<<<(TOK * 128 + 255) / 256, 256, 0, stream>>>(tk, rot, TOK * 128, 256);
  gemm(tk, B_k, nullptr, kh, 256, 256, 256, 256, 256, 0);

  // up projections
  gemm(qh, Wqu, bqu, qq, 2048, 1024, 1024, 1024, 2048, 1);
  gemm(kh, Wku, bku, kk,  512,  256,  256,  256,  512, 1);
  gemm(vh, Wvu, bvu, vv,  512,  256,  256,  256,  512, 1);

  // q[:, :1024] = q @ C_q (write low cols of qf, copy high cols)
  gemm(qq, C_q, nullptr, qf, 1024, 2048, 2048, 1024, 2048, 0);
  copycols_kernel<<<(TOK * 1024 + 255) / 256, 256, 0, stream>>>(qq, qf, 2048, 1024, 1024);
  // k[:, :256] = k @ C_k
  gemm(kk, C_k, nullptr, kf, 256, 512, 512, 256, 512, 0);
  copycols_kernel<<<(TOK * 256 + 255) / 256, 256, 0, stream>>>(kk, kf, 512, 256, 256);

  // flash attention: grid (S/16 q-tiles, 32 heads, B=2), one wave each
  dim3 ga(SEQ / 16, 32, 2);
  attn_kernel<<<ga, 32, 0, stream>>>(qf, kf, vv, at);

  // out = attn @ Wo^T
  gemm(at, Wo, nullptr, (float*)d_out, 2048, 2048, 2048, 2048, 2048, 1);
}
